// SequenceEncoder_35802847379793
// MI455X (gfx1250) — matrix operand order repaired and checked by reading, not compiled, not run on hardware
//
#include <hip/hip_runtime.h>
#include <hip/hip_bf16.h>

// ---------------------------------------------------------------------------
// SequenceEncoder for MI455X (gfx1250, wave32, WMMA + TDM).
//
//  * conv stack (93% of FLOPs) -> v_wmma_f32_16x16x32_f16 tiled GEMM:
//      y[:,l] = sum_{kk=0..8} W_kk(128x128) @ x[:, l+kk]
//    Weights pre-packed into per-lane WMMA A-fragment order (contiguous 32B
//    per lane). Activations stored [b][l][d] (d contiguous) so each lane's
//    B fragment is 16 contiguous halves in LDS.
//  * X tile staged into LDS with the Tensor Data Mover (tensor_load_to_lds),
//    synchronized with s_wait_tensorcnt + workgroup barrier.
//  * attention tail collapsed algebraically:
//      scores_l = (Wk^T q) . x_l / sqrt(d)   (+const dropped by softmax)
//      out      = Wv (sum_l attn_l x_l) + bv (sum attn = 1)
// ---------------------------------------------------------------------------

typedef __attribute__((ext_vector_type(16))) _Float16     v16h;
typedef __attribute__((ext_vector_type(8)))  _Float16     v8h;
typedef __attribute__((ext_vector_type(8)))  float        v8f;
typedef __attribute__((ext_vector_type(4)))  unsigned int v4u;
typedef __attribute__((ext_vector_type(8)))  unsigned int v8u;

#define BB 128
#define LL 4096
#define DD 128
#define KK 9
#define L1 4088
#define L2 4080
#define L3 4072
#define NT 14                     // 16-wide N tiles per wave
#define TILE_L (NT * 16)          // 224 output columns per workgroup
#define HALO 8
#define TILE_ROWS (TILE_L + HALO) // 232 rows staged (232*256B = 59392B LDS)
#define WFRAG_PER_LAYER (KK * 8 * 4 * 32 * 16)   // 147456 halves per layer

// ---------------------------------------------------------------------------
// Pack conv weights (f32, torch OIH [O][I][K]) into WMMA A-fragment order:
// wfrag[layer][kk][dout_blk][kchunk][lane][slot], 16 halves contiguous/lane.
// A 16x32 f16 fragment layout (contiguous, matches B): lane<16 -> M=lane, K in
// {0..15}; lane>=16 -> M=lane-16, K in {16..31}.
// ---------------------------------------------------------------------------
__global__ void pack_weights_kernel(const float* __restrict__ conv1_w,
                                    const float* __restrict__ conv_rest_w,
                                    _Float16* __restrict__ wfrag) {
  int idx = blockIdx.x * 256 + threadIdx.x;
  if (idx >= 3 * WFRAG_PER_LAYER) return;
  int s = idx & 15;
  int t = idx >> 4;
  int lane = t & 31;  t >>= 5;
  int kc   = t & 3;   t >>= 2;
  int db   = t & 7;   t >>= 3;
  int kk   = t % KK;
  int layer = t / KK;
  int m   = lane & 15;
  int grp = lane >> 4;
  int kloc = grp * 16 + s;
  int din  = kc * 32 + kloc;
  int dout = db * 16 + m;
  float w;
  if (layer == 0)
    w = conv1_w[(dout * DD + din) * KK + kk];
  else
    w = conv_rest_w[(((layer - 1) * DD + dout) * DD + din) * KK + kk];
  wfrag[idx] = (_Float16)w;
}

// ---------------------------------------------------------------------------
// Embedding gather -> f16 activations, layout [b][l][d] (d contiguous).
// ---------------------------------------------------------------------------
__global__ void embed_kernel(const int* __restrict__ tokens,
                             const float* __restrict__ emb,
                             _Float16* __restrict__ x0) {
  long idx = (long)blockIdx.x * 256 + threadIdx.x;   // over B*L*16
  if (idx >= (long)BB * LL * 16) return;
  int c = (int)(idx & 15);
  long bl = idx >> 4;
  int tok = tokens[bl];
  const float* e = emb + tok * DD + c * 8;
  v8h h;
#pragma unroll
  for (int g = 0; g < 8; ++g) h[g] = (_Float16)e[g];
  *(v8h*)(&x0[bl * DD + c * 8]) = h;
}

// ---------------------------------------------------------------------------
// WMMA conv layer. Workgroup = 8 wave32, computes 128 dout x 224 l for one
// batch. Wave w owns dout rows [16w,16w+16); NT=14 accumulator tiles across l.
// X tile staged to LDS by the Tensor Data Mover (one 2-D descriptor:
// rows of 32 x 8B elements, 232 rows).
// ---------------------------------------------------------------------------
__global__ __launch_bounds__(256) void conv_wmma_kernel(
    const _Float16* __restrict__ xin,
    const _Float16* __restrict__ wfrag,   // this layer's packed fragments
    const float* __restrict__ bias,
    _Float16* __restrict__ yout,
    int Lin, int Lout) {
  __shared__ _Float16 lds_x[TILE_ROWS * DD];   // 232*128*2 = 59392 B

  const int b   = blockIdx.y;
  const int l0  = blockIdx.x * TILE_L;
  const int tid = threadIdx.x;
  const int wave = tid >> 5;        // 0..7 (wave32!)
  const int lane = tid & 31;
  const int lane_lo = lane & 15;
  const int hi = lane >> 4;

  // ---- TDM: DMA the X tile (TILE_ROWS rows x 256B) into LDS. -------------
  if (wave == 0) {
    unsigned long long ga =
        (unsigned long long)(xin + ((size_t)b * Lin + l0) * DD);
    unsigned int lds_off = (unsigned int)(unsigned long long)(&lds_x[0]);
    unsigned int td0   = 32u;                 // row length in 8B elements
    unsigned int rows  = (unsigned int)Lin;   // tensor dim1 (for OOB zero)
    v4u g0;
    g0[0] = 1u;                                        // count=1 user D#
    g0[1] = lds_off;                                   // lds_addr
    g0[2] = (unsigned int)(ga & 0xFFFFFFFFu);          // global_addr[31:0]
    g0[3] = (unsigned int)((ga >> 32) & 0x01FFFFFFu)   // global_addr[56:32]
            | (2u << 30);                              // type=2 ("image")
    v8u g1;
    g1[0] = 0x00030000u;                               // data_size=3 (8B)
    g1[1] = (td0 & 0xFFFFu) << 16;                     // tensor_dim0 lo16
    g1[2] = (td0 >> 16) | ((rows & 0xFFFFu) << 16);    // td0 hi | tensor_dim1 lo
    g1[3] = (rows >> 16) | (32u << 16);                // td1 hi | tile_dim0=32
    g1[4] = (unsigned int)TILE_ROWS;                   // tile_dim1 | tile_dim2=0
    g1[5] = 32u;                                       // tensor_dim0_stride lo
    g1[6] = 0u;                                        // stride hi | dim1_stride lo
    g1[7] = 0u;
    v4u g2 = {0u, 0u, 0u, 0u};
    v4u g3 = {0u, 0u, 0u, 0u};
    asm volatile("tensor_load_to_lds %0, %1, %2, %3"
                 :: "s"(g0), "s"(g1), "s"(g2), "s"(g3)
                 : "memory");
    __builtin_amdgcn_s_wait_tensorcnt(0);
  }
  __syncthreads();

  v8f acc[NT] = {};

  for (int kk = 0; kk < KK; ++kk) {
    // Prefetch next shift's A fragments (global_prefetch_b8).
    __builtin_prefetch(wfrag + (((size_t)(kk + 1) * 8 + wave) * 4) * 32 * 16,
                       0, 0);
#pragma unroll
    for (int kc = 0; kc < 4; ++kc) {
      // A fragment: 32 contiguous bytes per lane from packed weights.
      const v16h* ap = (const v16h*)(wfrag +
          ((((size_t)kk * 8 + wave) * 4 + kc) * 32 + lane) * 16);
      v16h a = *ap;
#pragma unroll
      for (int nt = 0; nt < NT; ++nt) {
        // B fragment 32x16: lanes 0-15 -> K=0..15 (din kc*32..+15) at
        // col N=lane; lanes 16-31 -> K=16..31. 16 contiguous halves in LDS.
        const v16h* bp = (const v16h*)(&lds_x[(nt * 16 + lane_lo + kk) * DD +
                                              kc * 32 + hi * 16]);
        v16h bf = *bp;
        acc[nt] = __builtin_amdgcn_wmma_f32_16x16x32_f16(
            false, a, false, bf, (short)0, acc[nt], false, false);
      }
    }
  }

  // Store: D-fragment element g of lane -> M = g + 8*hi, N = lane_lo.
  // Lane's 8 outputs are contiguous in d -> single 16B store.
  const int dbase = wave * 16 + hi * 8;
  float bloc[8];
#pragma unroll
  for (int g = 0; g < 8; ++g) bloc[g] = bias[dbase + g];
#pragma unroll
  for (int nt = 0; nt < NT; ++nt) {
    int l = l0 + nt * 16 + lane_lo;
    if (l < Lout) {
      v8h h;
#pragma unroll
      for (int g = 0; g < 8; ++g) h[g] = (_Float16)(acc[nt][g] + bloc[g]);
      *(v8h*)(&yout[((size_t)b * Lout + l) * DD + dbase]) = h;
    }
  }
}

// ---------------------------------------------------------------------------
// read_mean over the sequence axis.
// ---------------------------------------------------------------------------
__global__ void mean_kernel(const _Float16* __restrict__ x,
                            float* __restrict__ rm, int Lp) {
  int b = blockIdx.x, d = threadIdx.x;
  const _Float16* xp = x + (size_t)b * Lp * DD + d;
  float s = 0.f;
  for (int l = 0; l < Lp; ++l) s += (float)xp[(size_t)l * DD];
  rm[b * DD + d] = s / (float)Lp;
}

// ---------------------------------------------------------------------------
// q = Wq rm + bq ; qk = Wk^T q, pre-scaled by 1/sqrt(d).
// (q . bk constant is dropped: softmax is shift-invariant.)
// ---------------------------------------------------------------------------
__global__ void qk_kernel(const float* __restrict__ rm,
                          const float* __restrict__ wq_w,
                          const float* __restrict__ wq_b,
                          const float* __restrict__ wk_w,
                          float* __restrict__ qk) {
  __shared__ float q[DD];
  __shared__ float r[DD];
  int b = blockIdx.x, d = threadIdx.x;
  r[d] = rm[b * DD + d];
  __syncthreads();
  float s = wq_b[d];
  for (int e = 0; e < DD; ++e) s += wq_w[d * DD + e] * r[e];
  q[d] = s;
  __syncthreads();
  float t = 0.f;
  for (int dd = 0; dd < DD; ++dd) t += q[dd] * wk_w[dd * DD + d];
  qk[b * DD + d] = t * 0.08838834764831845f;   // 1/sqrt(128)
}

// ---------------------------------------------------------------------------
// scores[b][l] = qk[b] . x[b][l]
// ---------------------------------------------------------------------------
__global__ void score_kernel(const _Float16* __restrict__ x,
                             const float* __restrict__ qk,
                             float* __restrict__ scores, int Lp) {
  long idx = (long)blockIdx.x * 256 + threadIdx.x;
  if (idx >= (long)BB * Lp) return;
  int b = (int)(idx / Lp);
  long l = idx % Lp;
  const _Float16* xp = x + ((size_t)b * Lp + l) * DD;
  const float* qp = qk + b * DD;
  float s = 0.f;
  for (int d = 0; d < DD; ++d) s += qp[d] * (float)xp[d];
  scores[idx] = s;
}

// ---------------------------------------------------------------------------
// softmax over Lp + attention-weighted pooling of x -> pooled[b][d].
// ---------------------------------------------------------------------------
__global__ void pool_kernel(const _Float16* __restrict__ x,
                            const float* __restrict__ scores,
                            float* __restrict__ pooled, int Lp) {
  __shared__ float red[DD];
  int b = blockIdx.x, tid = threadIdx.x;
  const float* sc = scores + (size_t)b * Lp;

  float m = -1e30f;
  for (int l = tid; l < Lp; l += DD) m = fmaxf(m, sc[l]);
  red[tid] = m;
  __syncthreads();
  for (int st = 64; st > 0; st >>= 1) {
    if (tid < st) red[tid] = fmaxf(red[tid], red[tid + st]);
    __syncthreads();
  }
  m = red[0];
  __syncthreads();

  float s = 0.f;
  for (int l = tid; l < Lp; l += DD) s += __expf(sc[l] - m);
  red[tid] = s;
  __syncthreads();
  for (int st = 64; st > 0; st >>= 1) {
    if (tid < st) red[tid] += red[tid + st];
    __syncthreads();
  }
  float denom = red[0];

  const _Float16* xp = x + (size_t)b * Lp * DD + tid;
  float acc = 0.f;
  for (int l = 0; l < Lp; ++l)
    acc += __expf(sc[l] - m) * (float)xp[(size_t)l * DD];
  pooled[b * DD + tid] = acc / denom;
}

// ---------------------------------------------------------------------------
// out = Wv pooled + bv  (sum of attention weights is 1).
// ---------------------------------------------------------------------------
__global__ void out_kernel(const float* __restrict__ pooled,
                           const float* __restrict__ wv_w,
                           const float* __restrict__ wv_b,
                           float* __restrict__ out) {
  __shared__ float p[DD];
  int b = blockIdx.x, d = threadIdx.x;
  p[d] = pooled[b * DD + d];
  __syncthreads();
  float s = wv_b[d];
  for (int e = 0; e < DD; ++e) s += wv_w[d * DD + e] * p[e];
  out[b * DD + d] = s;
}

// ---------------------------------------------------------------------------
extern "C" void kernel_launch(void* const* d_in, const int* in_sizes, int n_in,
                              void* d_out, int out_size, void* d_ws, size_t ws_size,
                              hipStream_t stream) {
  const int*   tokens      = (const int*)  d_in[0];
  const float* emb         = (const float*)d_in[1];
  const float* conv1_w     = (const float*)d_in[2];
  const float* conv1_b     = (const float*)d_in[3];
  const float* conv_rest_w = (const float*)d_in[4];
  const float* conv_rest_b = (const float*)d_in[5];
  const float* wq_w        = (const float*)d_in[6];
  const float* wq_b        = (const float*)d_in[7];
  const float* wk_w        = (const float*)d_in[8];
  // d_in[9] = wk_b: mathematically irrelevant (softmax shift invariance).
  const float* wv_w        = (const float*)d_in[10];
  const float* wv_b        = (const float*)d_in[11];

  // Workspace layout (ping-pong activation buffers + small tail buffers).
  const size_t BUF = (size_t)BB * LL * DD * sizeof(_Float16);  // 128 MiB
  _Float16* bufA  = (_Float16*)d_ws;
  _Float16* bufB  = (_Float16*)((char*)d_ws + BUF);
  _Float16* wfrag = (_Float16*)((char*)d_ws + 2 * BUF);
  float* rm     = (float*)((char*)d_ws + 2 * BUF + (1 << 20));
  float* qkb    = rm + BB * DD;
  float* pooled = qkb + BB * DD;
  float* scores = pooled + BB * DD;

  // 1. Pack conv weights into WMMA fragment order (f16).
  pack_weights_kernel<<<(3 * WFRAG_PER_LAYER + 255) / 256, 256, 0, stream>>>(
      conv1_w, conv_rest_w, wfrag);

  // 2. Embedding gather -> f16 [b][l][d].
  embed_kernel<<<(int)(((long)BB * LL * 16 + 255) / 256), 256, 0, stream>>>(
      tokens, emb, bufA);

  // 3. Conv stack (WMMA): 4096 -> 4088 -> 4080 -> 4072, ping-pong buffers.
  dim3 cgrid((L1 + TILE_L - 1) / TILE_L, BB);   // (19, 128)
  conv_wmma_kernel<<<cgrid, 256, 0, stream>>>(bufA, wfrag, conv1_b,
                                              bufB, LL, L1);
  conv_wmma_kernel<<<cgrid, 256, 0, stream>>>(bufB, wfrag + WFRAG_PER_LAYER,
                                              conv_rest_b, bufA, L1, L2);
  conv_wmma_kernel<<<cgrid, 256, 0, stream>>>(bufA, wfrag + 2 * WFRAG_PER_LAYER,
                                              conv_rest_b + DD, bufB, L2, L3);

  // 4. Attention pooling tail (algebraically collapsed).
  mean_kernel <<<BB, DD, 0, stream>>>(bufB, rm, L3);
  qk_kernel   <<<BB, DD, 0, stream>>>(rm, wq_w, wq_b, wk_w, qkb);
  score_kernel<<<(int)(((long)BB * L3 + 255) / 256), 256, 0, stream>>>(
      bufB, qkb, scores, L3);
  pool_kernel <<<BB, DD, 0, stream>>>(bufB, scores, pooled, L3);
  out_kernel  <<<BB, DD, 0, stream>>>(pooled, wv_w, wv_b, (float*)d_out);
}
